// AdaptiveAttentionBlock_76459007803726
// MI455X (gfx1250) — compile-verified, matrix operand order checked by reading, not run on hardware
//
#include <hip/hip_runtime.h>
#include <hip/hip_bf16.h>

// Problem constants
constexpr int B_  = 2;
constexpr int S_  = 2048;
constexpr int H_  = 1024;
constexpr int NH_ = 16;
constexpr int DH_ = 64;
constexpr int M_  = B_ * S_;   // 4096 rows of activations

typedef __attribute__((ext_vector_type(16))) __bf16 v16bf;
typedef __attribute__((ext_vector_type(8)))  __bf16 v8bf;
typedef __attribute__((ext_vector_type(8)))  float  v8f;

// ---------------------------------------------------------------------------
// Tile loaders matching CDNA5 WMMA 16-bit A/B VGPR layout (ISA 7.12.2):
// lane L (half g = L>>4) holds row/col L&15, K-values [8g,8g+8) ++ [16+8g,16+8g+8)
// ---------------------------------------------------------------------------
__device__ __forceinline__ v16bf ldtile_bf16(const __bf16* row, int k0, int g) {
    const __bf16* p = row + k0 + (g << 3);
    v8bf lo = *(const v8bf*)(p);
    v8bf hi = *(const v8bf*)(p + 16);
    v16bf r;
#pragma unroll
    for (int i = 0; i < 8; ++i) { r[i] = lo[i]; r[i + 8] = hi[i]; }
    return r;
}

#define WMMA_BF16(A, Bm, C) \
    __builtin_amdgcn_wmma_f32_16x16x32_bf16(false, (A), false, (Bm), (short)0, (C), false, false)

// ---------------------------------------------------------------------------
// f32 -> bf16 conversion
// ---------------------------------------------------------------------------
__global__ void f32_to_bf16(const float* __restrict__ in, __bf16* __restrict__ out, int n) {
    int i = blockIdx.x * blockDim.x + threadIdx.x;
    if (i < n) out[i] = (__bf16)in[i];
}

// ---------------------------------------------------------------------------
// Generic GEMM: C[m,n] = sum_k A[m,k] * W[n,k] + bias[n]  (i.e., A @ W^T + b)
// mode 0: store Q  bf16 [B,NH,S,DH], scaled by 1/sqrt(DH)
// mode 1: store K  bf16 [B,NH,S,DH]
// mode 2: store V  bf16 transposed [B,NH,DH,S]
// mode 3: store f32 res = C + resid (residual add), layout [M,H]
// Wave tile: 16 (M) x 64 (N); block = 4 waves covering 64x64.
// ---------------------------------------------------------------------------
__global__ __launch_bounds__(128)
void gemm16x64_bf16(const __bf16* __restrict__ A,
                    const __bf16* __restrict__ W,
                    const float*  __restrict__ bias,
                    const float*  __restrict__ resid,
                    __bf16* __restrict__ outBf,
                    float*  __restrict__ outF,
                    int Kdim, int mode) {
    const int lane = threadIdx.x & 31;
    const int wave = threadIdx.x >> 5;
    const int g    = lane >> 4;
    const int h16  = lane & 15;
    const int m0   = (blockIdx.x * 4 + wave) * 16;
    const int n0   = blockIdx.y * 64;

    v8f acc[4] = {};

    const __bf16* arow = A + (size_t)(m0 + h16) * Kdim;
    const __bf16* wrow[4];
#pragma unroll
    for (int j = 0; j < 4; ++j)
        wrow[j] = W + (size_t)(n0 + j * 16 + h16) * Kdim;

    for (int k0 = 0; k0 < Kdim; k0 += 32) {
        __builtin_prefetch(arow + k0 + 256, 0, 0);
        __builtin_prefetch(wrow[0] + k0 + 256, 0, 0);
        v16bf a = ldtile_bf16(arow, k0, g);
#pragma unroll
        for (int j = 0; j < 4; ++j) {
            v16bf b = ldtile_bf16(wrow[j], k0, g);
            acc[j] = WMMA_BF16(a, b, acc[j]);
        }
    }

    // C/D layout: lane L -> n = L&15 ; vgpr r -> m = r + 8*(L>>4)
#pragma unroll
    for (int j = 0; j < 4; ++j) {
        const int n = n0 + j * 16 + h16;
        const float bia = bias[n];
#pragma unroll
        for (int r = 0; r < 8; ++r) {
            const int m = m0 + r + 8 * g;
            float v = acc[j][r] + bia;
            if (mode == 0 || mode == 1) {
                const int b  = m >> 11;          // m / S_
                const int s  = m & (S_ - 1);
                const int nh = n >> 6;           // n / DH_
                const int dh = n & (DH_ - 1);
                if (mode == 0) v *= 0.125f;      // 1/sqrt(64)
                outBf[(((size_t)b * NH_ + nh) * S_ + s) * DH_ + dh] = (__bf16)v;
            } else if (mode == 2) {
                const int b  = m >> 11;
                const int s  = m & (S_ - 1);
                const int nh = n >> 6;
                const int dh = n & (DH_ - 1);
                outBf[(((size_t)b * NH_ + nh) * DH_ + dh) * S_ + s] = (__bf16)v;
            } else {
                const size_t idx = (size_t)m * H_ + n;
                outF[idx] = v + resid[idx];
            }
        }
    }
}

// ---------------------------------------------------------------------------
// Fused attention: per (bh, 64 q-rows) block, each wave owns 16 q-rows.
// Pass 1: scores via WMMA (Q tile kept in VGPRs), online row max / exp-sum.
// Pass 2: recompute scores, p = exp(s-max)/sum * attw[k]; write probs ONCE
//         with a non-temporal store (537MB stream, never re-read), stage the
//         bf16 p-tile in per-wave LDS, feed PV WMMA chain.
// ---------------------------------------------------------------------------
__global__ __launch_bounds__(128)
void attn_fused(const __bf16* __restrict__ Q,
                const __bf16* __restrict__ K,
                const __bf16* __restrict__ Vt,
                const int*   __restrict__ mask,
                const float* __restrict__ attw,
                float*  __restrict__ probs,
                __bf16* __restrict__ ctxb) {
    __shared__ __bf16 sp[4][16 * 64];   // 2KB per wave

    const int lane = threadIdx.x & 31;
    const int wave = threadIdx.x >> 5;
    const int g    = lane >> 4;
    const int h16  = lane & 15;
    const int bh   = blockIdx.z;
    const int b    = bh >> 4;               // bh / NH_
    const int h    = bh & 15;
    const int m0   = (blockIdx.x * 4 + wave) * 16;

    const __bf16* Qh = Q + (size_t)bh * S_ * DH_;
    const __bf16* Kh = K + (size_t)bh * S_ * DH_;

    // Q tile: 16 rows x 64 (full DH), held in registers for both passes
    const __bf16* arow = Qh + (size_t)(m0 + h16) * DH_;
    const v16bf aq0 = ldtile_bf16(arow, 0, g);
    const v16bf aq1 = ldtile_bf16(arow, 32, g);

    float mx[8], sm[8];
#pragma unroll
    for (int r = 0; r < 8; ++r) { mx[r] = -__builtin_inff(); sm[r] = 0.f; }

    // ---------------- pass 1: online max / sum over all keys ----------------
    for (int n0 = 0; n0 < S_; n0 += 64) {
        v8f acc[4] = {};
#pragma unroll
        for (int j = 0; j < 4; ++j) {
            const __bf16* brow = Kh + (size_t)(n0 + j * 16 + h16) * DH_;
            acc[j] = WMMA_BF16(aq0, ldtile_bf16(brow, 0,  g), acc[j]);
            acc[j] = WMMA_BF16(aq1, ldtile_bf16(brow, 32, g), acc[j]);
        }
#pragma unroll
        for (int j = 0; j < 4; ++j) {
            const int n = n0 + j * 16 + h16;
            const float maskadd = (1.0f - (float)mask[b * S_ + n]) * -10000.0f;
#pragma unroll
            for (int r = 0; r < 8; ++r) {
                const float v  = acc[j][r] + maskadd;
                const float nm = fmaxf(mx[r], v);
                sm[r] = sm[r] * __expf(mx[r] - nm) + __expf(v - nm);
                mx[r] = nm;
            }
        }
    }
    // Butterfly reduce (max,sum) across the 16 lanes holding each row
#pragma unroll
    for (int msk = 1; msk <= 8; msk <<= 1) {
#pragma unroll
        for (int r = 0; r < 8; ++r) {
            const float om = __shfl_xor(mx[r], msk, 32);
            const float os = __shfl_xor(sm[r], msk, 32);
            const float nm = fmaxf(mx[r], om);
            sm[r] = sm[r] * __expf(mx[r] - nm) + os * __expf(om - nm);
            mx[r] = nm;
        }
    }
    float inv[8];
#pragma unroll
    for (int r = 0; r < 8; ++r) inv[r] = 1.0f / sm[r];

    // ---------------- pass 2: probs write + PV accumulate -------------------
    v8f cacc[4] = {};
    __bf16* myTile = &sp[wave][0];

    for (int n0 = 0; n0 < S_; n0 += 64) {
        v8f acc[4] = {};
#pragma unroll
        for (int j = 0; j < 4; ++j) {
            const __bf16* brow = Kh + (size_t)(n0 + j * 16 + h16) * DH_;
            acc[j] = WMMA_BF16(aq0, ldtile_bf16(brow, 0,  g), acc[j]);
            acc[j] = WMMA_BF16(aq1, ldtile_bf16(brow, 32, g), acc[j]);
        }
#pragma unroll
        for (int j = 0; j < 4; ++j) {
            const int n = n0 + j * 16 + h16;
            const float maskadd = (1.0f - (float)mask[b * S_ + n]) * -10000.0f;
            const float aw = attw[b * S_ + n];
#pragma unroll
            for (int r = 0; r < 8; ++r) {
                const int m = m0 + r + 8 * g;
                const float p = __expf(acc[j][r] + maskadd - mx[r]) * inv[r] * aw;
                // Streaming NT store: written once, never re-read on device.
                __builtin_nontemporal_store(p, &probs[((size_t)bh * S_ + m) * S_ + n]);
                myTile[(r + 8 * g) * 64 + (j * 16 + h16)] = (__bf16)p;
            }
        }
        // PV: A from per-wave LDS tile (same wave wrote it; DS ops are in-order)
        const __bf16* prow = myTile + h16 * 64;
        const v16bf pa0 = ldtile_bf16(prow, 0,  g);
        const v16bf pa1 = ldtile_bf16(prow, 32, g);
#pragma unroll
        for (int j = 0; j < 4; ++j) {
            const __bf16* vrow = Vt + ((size_t)bh * DH_ + j * 16 + h16) * S_ + n0;
            cacc[j] = WMMA_BF16(pa0, ldtile_bf16(vrow, 0,  g), cacc[j]);
            cacc[j] = WMMA_BF16(pa1, ldtile_bf16(vrow, 32, g), cacc[j]);
        }
    }

    // ctx store: [B,S,H] bf16, heads re-interleaved for the O-projection
#pragma unroll
    for (int j = 0; j < 4; ++j) {
        const int d = j * 16 + h16;
#pragma unroll
        for (int r = 0; r < 8; ++r) {
            const int m = m0 + r + 8 * g;
            ctxb[((size_t)(b * S_ + m)) * H_ + h * DH_ + d] = (__bf16)cacc[j][r];
        }
    }
}

// ---------------------------------------------------------------------------
// LayerNorm over H=1024 per row. Row is staged into LDS with gfx1250
// async global->LDS copies (ASYNCcnt), then reduced. Output stream is NT.
// ---------------------------------------------------------------------------
__global__ __launch_bounds__(256)
void layernorm_rows(const float* __restrict__ res,
                    const float* __restrict__ gamma,
                    const float* __restrict__ beta,
                    float* __restrict__ out) {
    const size_t row = blockIdx.x;
    const float* r = res + row * (size_t)H_;
    const int t = threadIdx.x;

    __shared__ float srow[H_];
    __shared__ float s1[256];
    __shared__ float s2[256];

    // Stage the row into LDS via async copies. Generic LDS pointers carry the
    // LDS byte address in their low 32 bits (ISA 10.2), so truncation is valid.
#pragma unroll
    for (int i = 0; i < 4; ++i) {
        const int c = t + i * 256;
        unsigned lds_addr = (unsigned)(size_t)(&srow[c]);
        unsigned long long ga = (unsigned long long)(size_t)(r + c);
        asm volatile("global_load_async_to_lds_b32 %0, %1, off"
                     :: "v"(lds_addr), "v"(ga) : "memory");
    }
    asm volatile("s_wait_asynccnt 0x0" ::: "memory");
    __syncthreads();

    float vals[4];
    float sum = 0.f, sq = 0.f;
#pragma unroll
    for (int i = 0; i < 4; ++i) {
        vals[i] = srow[t + i * 256];
        sum += vals[i];
        sq  += vals[i] * vals[i];
    }
    s1[t] = sum; s2[t] = sq;
    __syncthreads();
    for (int s3 = 128; s3 > 0; s3 >>= 1) {
        if (t < s3) { s1[t] += s1[t + s3]; s2[t] += s2[t + s3]; }
        __syncthreads();
    }
    const float mu  = s1[0] * (1.0f / (float)H_);
    const float var = s2[0] * (1.0f / (float)H_) - mu * mu;
    const float rs  = rsqrtf(var + 1e-12f);

#pragma unroll
    for (int i = 0; i < 4; ++i) {
        const int c = t + i * 256;
        const float o = (vals[i] - mu) * rs * gamma[c] + beta[c];
        __builtin_nontemporal_store(o, &out[row * (size_t)H_ + c]);
    }
}

// ---------------------------------------------------------------------------
// Host launch
// ---------------------------------------------------------------------------
extern "C" void kernel_launch(void* const* d_in, const int* in_sizes, int n_in,
                              void* d_out, int out_size, void* d_ws, size_t ws_size,
                              hipStream_t stream) {
    (void)in_sizes; (void)n_in; (void)out_size; (void)ws_size;

    const float* X    = (const float*)d_in[0];
    const int*   mask = (const int*)d_in[1];
    const float* attw = (const float*)d_in[2];
    const float* Wq   = (const float*)d_in[3];
    const float* bq   = (const float*)d_in[4];
    const float* Wk   = (const float*)d_in[5];
    const float* bk   = (const float*)d_in[6];
    const float* Wv   = (const float*)d_in[7];
    const float* bv   = (const float*)d_in[8];
    const float* Wo   = (const float*)d_in[9];
    const float* bo   = (const float*)d_in[10];
    const float* gamma = (const float*)d_in[11];
    const float* beta  = (const float*)d_in[12];

    float* out_normed = (float*)d_out;
    float* probs      = (float*)d_out + (size_t)B_ * S_ * H_;

    // Workspace carve-out (all 256B aligned)
    char* ws = (char*)d_ws;
    size_t off = 0;
    auto carve = [&](size_t bytes) -> void* {
        void* p = ws + off;
        off += (bytes + 255) & ~(size_t)255;
        return p;
    };
    __bf16* Xb   = (__bf16*)carve((size_t)M_ * H_ * 2);
    __bf16* Wqb  = (__bf16*)carve((size_t)H_ * H_ * 2);
    __bf16* Wkb  = (__bf16*)carve((size_t)H_ * H_ * 2);
    __bf16* Wvb  = (__bf16*)carve((size_t)H_ * H_ * 2);
    __bf16* Wob  = (__bf16*)carve((size_t)H_ * H_ * 2);
    __bf16* Qb   = (__bf16*)carve((size_t)B_ * NH_ * S_ * DH_ * 2);
    __bf16* Kb   = (__bf16*)carve((size_t)B_ * NH_ * S_ * DH_ * 2);
    __bf16* Vtb  = (__bf16*)carve((size_t)B_ * NH_ * S_ * DH_ * 2);
    __bf16* ctxb = (__bf16*)carve((size_t)M_ * H_ * 2);
    float*  res  = (float*)carve((size_t)M_ * H_ * 4);

    // 1) convert activations + weights to bf16
    f32_to_bf16<<<(M_ * H_ + 255) / 256, 256, 0, stream>>>(X, Xb, M_ * H_);
    f32_to_bf16<<<(H_ * H_ + 255) / 256, 256, 0, stream>>>(Wq, Wqb, H_ * H_);
    f32_to_bf16<<<(H_ * H_ + 255) / 256, 256, 0, stream>>>(Wk, Wkb, H_ * H_);
    f32_to_bf16<<<(H_ * H_ + 255) / 256, 256, 0, stream>>>(Wv, Wvb, H_ * H_);
    f32_to_bf16<<<(H_ * H_ + 255) / 256, 256, 0, stream>>>(Wo, Wob, H_ * H_);

    // 2) QKV projections (WMMA)
    dim3 blk(128);
    dim3 grdP(M_ / 64, H_ / 64);
    gemm16x64_bf16<<<grdP, blk, 0, stream>>>(Xb, Wqb, bq, nullptr, Qb,  nullptr, H_, 0);
    gemm16x64_bf16<<<grdP, blk, 0, stream>>>(Xb, Wkb, bk, nullptr, Kb,  nullptr, H_, 1);
    gemm16x64_bf16<<<grdP, blk, 0, stream>>>(Xb, Wvb, bv, nullptr, Vtb, nullptr, H_, 2);

    // 3) fused scores + softmax + attw scaling + PV (single NT probs write)
    dim3 grdA(S_ / 64, 1, B_ * NH_);
    attn_fused<<<grdA, blk, 0, stream>>>(Qb, Kb, Vtb, mask, attw, probs, ctxb);

    // 4) out-projection + residual (WMMA)
    gemm16x64_bf16<<<grdP, blk, 0, stream>>>(ctxb, Wob, bo, X, nullptr, res, H_, 3);

    // 5) layernorm (async global->LDS staging, NT output)
    layernorm_rows<<<dim3(M_), 256, 0, stream>>>(res, gamma, beta, out_normed);
}
